// VisionBDHv3_16595753632195
// MI455X (gfx1250) — matrix operand order repaired and checked by reading, not compiled, not run on hardware
//
#include <hip/hip_runtime.h>
#include <hip/hip_bf16.h>
#include <math.h>

// ---------------- model constants ----------------
#define BATCH 16
#define DMODEL 768
#define NHEAD 12
#define NLAT 256            // per-head latent dim
#define NLAYER 12
#define TSEQ 196            // real sequence length (14x14 patches)
#define TPAD 224            // padded to 14*16 (= 7*32, WMMA-exact in M/N/K)
#define GRIDP 14
#define NCLS 1000
#define BT (BATCH * TSEQ)   // 3136
#define HN (NHEAD * NLAT)   // 3072

typedef __bf16 bf16_t;
typedef __attribute__((ext_vector_type(16))) __bf16 v16bf;
typedef __attribute__((ext_vector_type(8)))  __bf16 v8bf;
typedef __attribute__((ext_vector_type(8)))  float  v8f;

#define WMMA_BF16(a, b, c) \
  __builtin_amdgcn_wmma_f32_16x16x32_bf16(false, (a), false, (b), (short)0, (c), false, false)

// ---------------- scalar bf16 helpers (bit-exact RNE) ----------------
__device__ __forceinline__ bf16_t tobf(float f) {
  unsigned u = __float_as_uint(f);
  unsigned r = (u + 0x7FFFu + ((u >> 16) & 1u)) >> 16;
  unsigned short s = (unsigned short)r;
  return __builtin_bit_cast(bf16_t, s);
}
__device__ __forceinline__ float frombf(bf16_t b) {
  unsigned short s = __builtin_bit_cast(unsigned short, b);
  return __uint_as_float(((unsigned)s) << 16);
}

// ---------------- WMMA fragment helpers (ISA 7.12.2 layouts, wave32) ------
__device__ __forceinline__ int lane_id() { return (int)(threadIdx.x & 31u); }

// A: 16x32 bf16, row-major source, lda in elements.
// lanes 0-15: row M=lane, K chunks {0..7} and {16..23};
// lanes 16-31: row M=lane-16, K chunks {8..15} and {24..31}.
__device__ __forceinline__ v16bf load_a_frag(const bf16_t* A, int lda) {
  int lane = lane_id();
  const bf16_t* p = A + (size_t)(lane & 15) * lda + ((lane >> 4) << 3);
  v8bf lo = *(const v8bf*)p;
  v8bf hi = *(const v8bf*)(p + 16);
  return __builtin_shufflevector(lo, hi, 0, 1, 2, 3, 4, 5, 6, 7,
                                         8, 9, 10, 11, 12, 13, 14, 15);
}

// B: 32x16 bf16, row-major [K,N] source; lane L holds column N=L%16,
// K chunks {8h..8h+7} and {16+8h..23+8h}, h = L/16.
__device__ __forceinline__ v16bf load_b_frag(const bf16_t* B, int ldb) {
  int lane = lane_id();
  const bf16_t* p = B + (size_t)((lane >> 4) << 3) * ldb + (lane & 15);
  v16bf b;
#pragma unroll
  for (int s = 0; s < 16; ++s) b[s] = p[(size_t)(s < 8 ? s : s + 8) * ldb];
  return b;
}

// C/D f32 16x16: lanes 0-15 -> N=lane, VGPR v -> M=v; lanes 16-31 -> M=v+8.
__device__ __forceinline__ v8f load_c_frag(const float* C, int ldc) {
  int lane = lane_id();
  int n = lane & 15, m0 = (lane >> 4) << 3;
  v8f c;
#pragma unroll
  for (int i = 0; i < 8; ++i) c[i] = C[(size_t)(m0 + i) * ldc + n];
  return c;
}
__device__ __forceinline__ void store_d_f32(float* D, int ldc, v8f acc) {
  int lane = lane_id();
  int n = lane & 15, m0 = (lane >> 4) << 3;
#pragma unroll
  for (int i = 0; i < 8; ++i) D[(size_t)(m0 + i) * ldc + n] = acc[i];
}
__device__ __forceinline__ void store_d_bf16(bf16_t* D, int ldc, v8f acc, bool relu) {
  int lane = lane_id();
  int n = lane & 15, m0 = (lane >> 4) << 3;
#pragma unroll
  for (int i = 0; i < 8; ++i) {
    float v = acc[i];
    if (relu) v = fmaxf(v, 0.0f);
    D[(size_t)(m0 + i) * ldc + n] = tobf(v);
  }
}

// K-loop with one A fragment reused across NT B-tiles (16 x NT*16 output
// strip per wave). Prefetch of the next K block rides the gfx1250
// global_prefetch path.
template <int NT>
__device__ __forceinline__ void wmma_loop(const bf16_t* __restrict__ Ap, int lda,
                                          const bf16_t* __restrict__ Bp, int ldb,
                                          int K, v8f* acc) {
#pragma unroll 2
  for (int k = 0; k < K; k += 32) {
    if (k + 32 < K) {
      __builtin_prefetch(Ap + k + 32, 0, 1);
      __builtin_prefetch(Bp + (size_t)(k + 32) * ldb, 0, 1);
    }
    v16bf a = load_a_frag(Ap + k, lda);
#pragma unroll
    for (int j = 0; j < NT; ++j) {
      v16bf b = load_b_frag(Bp + (size_t)k * ldb + j * 16, ldb);
      acc[j] = WMMA_BF16(a, b, acc[j]);
    }
  }
}

// ---------------- elementwise / prep kernels ----------------

// im2patch + f32->bf16: xp[b*T+t][c*256+py*16+px] = x[b][c][gy*16+py][gx*16+px]
__global__ void k_patchify(const float* __restrict__ x, bf16_t* __restrict__ xp) {
  size_t i = (size_t)blockIdx.x * blockDim.x + threadIdx.x;
  if (i >= (size_t)BT * DMODEL) return;
  int m = (int)(i / DMODEL), j = (int)(i % DMODEL);
  int b = m / TSEQ, t = m % TSEQ;
  int c = j / 256, r = j % 256;
  int py = r / 16, px = r % 16;
  int gy = t / GRIDP, gx = t % GRIDP;
  size_t src = (((size_t)b * 3 + c) * 224 + gy * 16 + py) * 224 + gx * 16 + px;
  xp[i] = tobf(x[src]);
}

// patch_w [D_out, 768_in] -> B matrix [768_in, D_out] bf16
__global__ void k_transpose_pw(const float* __restrict__ w, bf16_t* __restrict__ wt) {
  size_t i = (size_t)blockIdx.x * blockDim.x + threadIdx.x;
  if (i >= (size_t)DMODEL * DMODEL) return;
  int k = (int)(i / DMODEL), n = (int)(i % DMODEL);
  wt[i] = tobf(w[(size_t)n * DMODEL + k]);
}

__global__ void k_cvt_bf16(const float* __restrict__ s, bf16_t* __restrict__ d, size_t n) {
  size_t i = (size_t)blockIdx.x * blockDim.x + threadIdx.x;
  if (i < n) d[i] = tobf(s[i]);
}

// RoPE tables for t in [0,TPAD), n in [0,256). THETA = 2^16.
__global__ void k_rope_tables(float* __restrict__ cosT, float* __restrict__ sinT) {
  int i = blockIdx.x * blockDim.x + threadIdx.x;
  if (i >= TPAD * NLAT) return;
  int t = i >> 8, n = i & 255;
  float e = (float)(2 * (n >> 1)) / (float)NLAT;
  float freq = exp2f(-16.0f * e);
  float ph = fmodf((float)t * freq, 1.0f) * 6.2831853071795864769f;
  cosT[i] = cosf(ph);
  sinT[i] = sinf(ph);
}

// LayerNorm row kernel: writes scaled bf16 rows into padded xn [B, TPAD, D]
__global__ void k_ln(const float* __restrict__ H, const float* __restrict__ w,
                     const float* __restrict__ bsh, float scale,
                     bf16_t* __restrict__ xn) {
  int row = blockIdx.x;                 // 0..BT-1
  int b = row / TSEQ, t = row % TSEQ;
  const float* src = H + (size_t)row * DMODEL;
  __shared__ float red[256];
  float s = 0.f;
  for (int j = threadIdx.x; j < DMODEL; j += 256) s += src[j];
  red[threadIdx.x] = s; __syncthreads();
  for (int k = 128; k > 0; k >>= 1) {
    if ((int)threadIdx.x < k) red[threadIdx.x] += red[threadIdx.x + k];
    __syncthreads();
  }
  float mu = red[0] / (float)DMODEL;
  __syncthreads();
  float v = 0.f;
  for (int j = threadIdx.x; j < DMODEL; j += 256) {
    float d = src[j] - mu; v += d * d;
  }
  red[threadIdx.x] = v; __syncthreads();
  for (int k = 128; k > 0; k >>= 1) {
    if ((int)threadIdx.x < k) red[threadIdx.x] += red[threadIdx.x + k];
    __syncthreads();
  }
  float rsig = rsqrtf(red[0] / (float)DMODEL + 1e-5f);
  bf16_t* dst = xn + ((size_t)b * TPAD + t) * DMODEL;
  for (int j = threadIdx.x; j < DMODEL; j += 256)
    dst[j] = tobf(((src[j] - mu) * rsig * w[j] + bsh[j]) * scale);
}

// qr = xs*cos + rot_half(xs)*sin over [B*NH, TPAD, NLAT].
// Also materializes qrt[bh][n][t] (transposed) so the scores GEMM can use
// contiguous 32-byte B-fragment loads instead of strided gathers.
__global__ void k_rope_apply(const bf16_t* __restrict__ xs,
                             const float* __restrict__ cosT,
                             const float* __restrict__ sinT,
                             bf16_t* __restrict__ qr,
                             bf16_t* __restrict__ qrt) {
  size_t i = (size_t)blockIdx.x * blockDim.x + threadIdx.x;
  size_t total = (size_t)BATCH * NHEAD * TPAD * NLAT;
  if (i >= total) return;
  int n = (int)(i & 255u);
  size_t rem = i >> 8;
  int t = (int)(rem % TPAD);
  int bh = (int)(rem / TPAD);
  float v  = frombf(xs[i]);
  float vp = frombf(xs[i ^ 1]);
  float rot = (n & 1) ? vp : -vp;
  int tn = t * NLAT + n;
  bf16_t r = tobf(v * cosT[tn] + rot * sinT[tn]);
  qr[i] = r;
  qrt[((size_t)bh * NLAT + n) * TPAD + t] = r;
}

// ---------------- WMMA GEMM kernels (1 wave = 16 x NT*16 strip) -----------

// h = xp @ pw^T + bias + pos : M=BT(196 tiles), N=768 (12 groups of 64)
__global__ void k_gemm_patch(const bf16_t* __restrict__ A, const bf16_t* __restrict__ B,
                             const float* __restrict__ bias, const float* __restrict__ pos,
                             float* __restrict__ H) {
  const int NG = DMODEL / 64;
  int strip = blockIdx.x * 4 + (threadIdx.x >> 5);
  if (strip >= (BT / 16) * NG) return;
  int tm = strip / NG, tg = strip % NG;
  const bf16_t* Ap = A + (size_t)tm * 16 * DMODEL;
  const bf16_t* Bp = B + tg * 64;
  v8f acc[4] = {};
  wmma_loop<4>(Ap, DMODEL, Bp, DMODEL, DMODEL, acc);
  int lane = lane_id();
  int m0 = tm * 16 + ((lane >> 4) << 3);
#pragma unroll
  for (int j = 0; j < 4; ++j) {
    int n = tg * 64 + j * 16 + (lane & 15);
#pragma unroll
    for (int i = 0; i < 8; ++i) {
      int m = m0 + i, t = m % TSEQ;
      H[(size_t)m * DMODEL + n] = acc[j][i] + bias[n] + pos[(size_t)t * DMODEL + n];
    }
  }
}

// xs[bh] = relu(xn[b] @ enc[h]) : per (b,h) M=TPAD(14), N=256 (4 groups)
__global__ void k_gemm_xs(const bf16_t* __restrict__ xn, const bf16_t* __restrict__ enc,
                          bf16_t* __restrict__ xs) {
  int bh = blockIdx.y;
  int b = bh / NHEAD, h = bh % NHEAD;
  const int NG = NLAT / 64;
  int strip = blockIdx.x * 4 + (threadIdx.x >> 5);
  if (strip >= (TPAD / 16) * NG) return;
  int tm = strip / NG, tg = strip % NG;
  const bf16_t* Ap = xn + ((size_t)b * TPAD + tm * 16) * DMODEL;
  const bf16_t* Bp = enc + (size_t)h * DMODEL * NLAT + tg * 64;
  v8f acc[4] = {};
  wmma_loop<4>(Ap, DMODEL, Bp, NLAT, DMODEL, acc);
  bf16_t* Dp = xs + (size_t)bh * TPAD * NLAT + (size_t)tm * 16 * NLAT + tg * 64;
#pragma unroll
  for (int j = 0; j < 4; ++j) store_d_bf16(Dp + j * 16, NLAT, acc[j], true);
}

// scores[bh] = qr[bh] @ qr[bh]^T : M=TPAD(14), N=TPAD (7 groups of 32), K=256
__global__ void k_gemm_scores(const bf16_t* __restrict__ qr,
                              const bf16_t* __restrict__ qrt,
                              bf16_t* __restrict__ sc) {
  int bh = blockIdx.y;
  const int NG = TPAD / 32;
  int strip = blockIdx.x * 2 + (threadIdx.x >> 5);  // 2 waves per block
  if (strip >= (TPAD / 16) * NG) return;
  int tm = strip / NG, tg = strip % NG;
  const bf16_t* Ap = qr + (size_t)bh * TPAD * NLAT + (size_t)tm * 16 * NLAT;
  const bf16_t* Bp = qrt + (size_t)bh * NLAT * TPAD + tg * 32;
  v8f acc[2] = {};
  wmma_loop<2>(Ap, NLAT, Bp, TPAD, NLAT, acc);
  bf16_t* Dp = sc + (size_t)bh * TPAD * TPAD + (size_t)tm * 16 * TPAD + tg * 32;
#pragma unroll
  for (int j = 0; j < 2; ++j) store_d_bf16(Dp + j * 16, TPAD, acc[j], false);
}

// y[bh] = scores[bh] @ xn[b] : M=TPAD(14), N=768 (12 groups), K=TPAD
__global__ void k_gemm_y(const bf16_t* __restrict__ sc, const bf16_t* __restrict__ xn,
                         bf16_t* __restrict__ y) {
  int bh = blockIdx.y;
  int b = bh / NHEAD;
  const int NG = DMODEL / 64;
  int strip = blockIdx.x * 4 + (threadIdx.x >> 5);
  if (strip >= (TPAD / 16) * NG) return;
  int tm = strip / NG, tg = strip % NG;
  const bf16_t* Ap = sc + (size_t)bh * TPAD * TPAD + (size_t)tm * 16 * TPAD;
  const bf16_t* Bp = xn + (size_t)b * TPAD * DMODEL + tg * 64;
  v8f acc[4] = {};
  wmma_loop<4>(Ap, TPAD, Bp, DMODEL, TPAD, acc);
  bf16_t* Dp = y + (size_t)bh * TPAD * DMODEL + (size_t)tm * 16 * DMODEL + tg * 64;
#pragma unroll
  for (int j = 0; j < 4; ++j) store_d_bf16(Dp + j * 16, DMODEL, acc[j], false);
}

// ys = relu(y[bh] @ encv[h]); g[(b*T+t)][h*256+n] = ys * xs  (t < TSEQ)
__global__ void k_gemm_ysgate(const bf16_t* __restrict__ y, const bf16_t* __restrict__ encv,
                              const bf16_t* __restrict__ xs, bf16_t* __restrict__ g) {
  int bh = blockIdx.y;
  int b = bh / NHEAD, h = bh % NHEAD;
  const int NG = NLAT / 64;
  int strip = blockIdx.x * 4 + (threadIdx.x >> 5);
  if (strip >= (TPAD / 16) * NG) return;
  int tm = strip / NG, tg = strip % NG;
  const bf16_t* Ap = y + (size_t)bh * TPAD * DMODEL + (size_t)tm * 16 * DMODEL;
  const bf16_t* Bp = encv + (size_t)h * DMODEL * NLAT + tg * 64;
  v8f acc[4] = {};
  wmma_loop<4>(Ap, DMODEL, Bp, NLAT, DMODEL, acc);
  int lane = lane_id();
  int m0 = tm * 16 + ((lane >> 4) << 3);
#pragma unroll
  for (int j = 0; j < 4; ++j) {
    int nn = tg * 64 + j * 16 + (lane & 15);
#pragma unroll
    for (int i = 0; i < 8; ++i) {
      int t = m0 + i;
      if (t < TSEQ) {
        float xv = frombf(xs[(size_t)bh * TPAD * NLAT + (size_t)t * NLAT + nn]);
        float val = fmaxf(acc[j][i], 0.0f) * xv;
        g[((size_t)(b * TSEQ + t)) * HN + h * NLAT + nn] = tobf(val);
      }
    }
  }
}

// h += g @ dec : M=BT(196), N=768 (12 groups), K=3072, C loaded from h
__global__ void k_gemm_out(const bf16_t* __restrict__ g, const bf16_t* __restrict__ dec,
                           float* __restrict__ H) {
  const int NG = DMODEL / 64;
  int strip = blockIdx.x * 4 + (threadIdx.x >> 5);
  if (strip >= (BT / 16) * NG) return;
  int tm = strip / NG, tg = strip % NG;
  const bf16_t* Ap = g + (size_t)tm * 16 * HN;
  const bf16_t* Bp = dec + tg * 64;
  float* Cp = H + (size_t)tm * 16 * DMODEL + tg * 64;
  v8f acc[4];
#pragma unroll
  for (int j = 0; j < 4; ++j) acc[j] = load_c_frag(Cp + j * 16, DMODEL);
  wmma_loop<4>(Ap, HN, Bp, DMODEL, HN, acc);
#pragma unroll
  for (int j = 0; j < 4; ++j) store_d_f32(Cp + j * 16, DMODEL, acc[j]);
}

// ---------------- tail kernels ----------------
__global__ void k_stats(const float* __restrict__ H, float2* __restrict__ st) {
  int row = blockIdx.x;
  const float* src = H + (size_t)row * DMODEL;
  __shared__ float red[256];
  float s = 0.f;
  for (int j = threadIdx.x; j < DMODEL; j += 256) s += src[j];
  red[threadIdx.x] = s; __syncthreads();
  for (int k = 128; k > 0; k >>= 1) {
    if ((int)threadIdx.x < k) red[threadIdx.x] += red[threadIdx.x + k];
    __syncthreads();
  }
  float mu = red[0] / (float)DMODEL;
  __syncthreads();
  float v = 0.f;
  for (int j = threadIdx.x; j < DMODEL; j += 256) {
    float d = src[j] - mu; v += d * d;
  }
  red[threadIdx.x] = v; __syncthreads();
  for (int k = 128; k > 0; k >>= 1) {
    if ((int)threadIdx.x < k) red[threadIdx.x] += red[threadIdx.x + k];
    __syncthreads();
  }
  if (threadIdx.x == 0) {
    float2 o; o.x = mu; o.y = rsqrtf(red[0] / (float)DMODEL + 1e-5f);
    st[row] = o;
  }
}

__global__ void k_pool(const float* __restrict__ H, const float2* __restrict__ st,
                       const float* __restrict__ w, const float* __restrict__ bb,
                       float* __restrict__ pooled) {
  int i = blockIdx.x * blockDim.x + threadIdx.x;
  if (i >= BATCH * DMODEL) return;
  int b = i / DMODEL, d = i % DMODEL;
  const float scale = 0.2773500981126146f;  // 1/sqrt(13)
  float s = 0.f;
  for (int t = 0; t < TSEQ; ++t) {
    int r = b * TSEQ + t;
    float2 m = st[r];
    s += (H[(size_t)r * DMODEL + d] - m.x) * m.y * w[d] + bb[d];
  }
  pooled[i] = s * scale / (float)TSEQ;
}

__global__ void k_head(const float* __restrict__ pooled, const float* __restrict__ hw,
                       const float* __restrict__ hb, float* __restrict__ out) {
  int i = blockIdx.x * blockDim.x + threadIdx.x;
  if (i >= BATCH * NCLS) return;
  int b = i / NCLS, c = i % NCLS;
  float s = hb[c];
  const float* p = pooled + (size_t)b * DMODEL;
#pragma unroll 4
  for (int d = 0; d < DMODEL; ++d) s += p[d] * hw[(size_t)d * NCLS + c];
  out[i] = s;
}

// ---------------- host orchestration ----------------
static inline size_t ws_alloc(size_t* off, size_t bytes) {
  size_t o = *off;
  *off = (o + bytes + 255) & ~(size_t)255;
  return o;
}

extern "C" void kernel_launch(void* const* d_in, const int* in_sizes, int n_in,
                              void* d_out, int out_size, void* d_ws, size_t ws_size,
                              hipStream_t stream) {
  (void)in_sizes; (void)n_in; (void)out_size; (void)ws_size;
  const float* x       = (const float*)d_in[0];
  const float* patch_w = (const float*)d_in[1];
  const float* patch_b = (const float*)d_in[2];
  const float* pos     = (const float*)d_in[3];
  const float* ln_w    = (const float*)d_in[4];
  const float* ln_b    = (const float*)d_in[5];
  const float* lnf_w   = (const float*)d_in[6];
  const float* lnf_b   = (const float*)d_in[7];
  const float* enc_f   = (const float*)d_in[8];
  const float* encv_f  = (const float*)d_in[9];
  const float* dec_f   = (const float*)d_in[10];
  const float* head_w  = (const float*)d_in[11];
  const float* head_b  = (const float*)d_in[12];
  float* out = (float*)d_out;

  char* ws = (char*)d_ws;
  size_t off = 0;
  bf16_t* XP   = (bf16_t*)(ws + ws_alloc(&off, (size_t)BT * DMODEL * 2));
  bf16_t* PW   = (bf16_t*)(ws + ws_alloc(&off, (size_t)DMODEL * DMODEL * 2));
  bf16_t* ENC  = (bf16_t*)(ws + ws_alloc(&off, (size_t)NHEAD * DMODEL * NLAT * 2));
  bf16_t* ENCV = (bf16_t*)(ws + ws_alloc(&off, (size_t)NHEAD * DMODEL * NLAT * 2));
  bf16_t* DEC  = (bf16_t*)(ws + ws_alloc(&off, (size_t)HN * DMODEL * 2));
  float*  H    = (float*) (ws + ws_alloc(&off, (size_t)BT * DMODEL * 4));
  bf16_t* XN   = (bf16_t*)(ws + ws_alloc(&off, (size_t)BATCH * TPAD * DMODEL * 2));
  bf16_t* XS   = (bf16_t*)(ws + ws_alloc(&off, (size_t)BATCH * NHEAD * TPAD * NLAT * 2));
  bf16_t* QR   = (bf16_t*)(ws + ws_alloc(&off, (size_t)BATCH * NHEAD * TPAD * NLAT * 2));
  bf16_t* QRT  = (bf16_t*)(ws + ws_alloc(&off, (size_t)BATCH * NHEAD * NLAT * TPAD * 2));
  bf16_t* SC   = (bf16_t*)(ws + ws_alloc(&off, (size_t)BATCH * NHEAD * TPAD * TPAD * 2));
  bf16_t* Y    = (bf16_t*)(ws + ws_alloc(&off, (size_t)BATCH * NHEAD * TPAD * DMODEL * 2));
  bf16_t* G    = (bf16_t*)(ws + ws_alloc(&off, (size_t)BT * HN * 2));
  float*  COS  = (float*) (ws + ws_alloc(&off, (size_t)TPAD * NLAT * 4));
  float*  SIN  = (float*) (ws + ws_alloc(&off, (size_t)TPAD * NLAT * 4));
  float2* ST   = (float2*)(ws + ws_alloc(&off, (size_t)BT * 8));
  float*  POOL = (float*) (ws + ws_alloc(&off, (size_t)BATCH * DMODEL * 4));

  // padded xn rows (t >= TSEQ) must be zero; zero once per call (capture-safe)
  hipMemsetAsync(XN, 0, (size_t)BATCH * TPAD * DMODEL * 2, stream);

  // ---- prep ----
  {
    size_t n = (size_t)BT * DMODEL;
    k_patchify<<<dim3((unsigned)((n + 255) / 256)), dim3(256), 0, stream>>>(x, XP);
  }
  {
    size_t n = (size_t)DMODEL * DMODEL;
    k_transpose_pw<<<dim3((unsigned)((n + 255) / 256)), dim3(256), 0, stream>>>(patch_w, PW);
  }
  {
    size_t n = (size_t)NHEAD * DMODEL * NLAT;
    unsigned g = (unsigned)((n + 255) / 256);
    k_cvt_bf16<<<dim3(g), dim3(256), 0, stream>>>(enc_f, ENC, n);
    k_cvt_bf16<<<dim3(g), dim3(256), 0, stream>>>(encv_f, ENCV, n);
    k_cvt_bf16<<<dim3(g), dim3(256), 0, stream>>>(dec_f, DEC, n);
  }
  k_rope_tables<<<dim3((TPAD * NLAT + 255) / 256), dim3(256), 0, stream>>>(COS, SIN);

  // ---- patch embed GEMM: 196 M-tiles * 12 N-groups = 2352 waves ----
  k_gemm_patch<<<dim3((BT / 16) * (DMODEL / 64) / 4), dim3(128), 0, stream>>>(
      XP, PW, patch_b, pos, H);

  // ---- layers (shared weights; per-layer LN + scale) ----
  for (int l = 0; l < NLAYER; ++l) {
    float scale = 1.0f / sqrtf((float)(l + 1));
    k_ln<<<dim3(BT), dim3(256), 0, stream>>>(H, ln_w + l * DMODEL, ln_b + l * DMODEL,
                                             scale, XN);
    k_gemm_xs<<<dim3((TPAD / 16) * (NLAT / 64) / 4, BATCH * NHEAD), dim3(128), 0, stream>>>(
        XN, ENC, XS);
    {
      size_t n = (size_t)BATCH * NHEAD * TPAD * NLAT;
      k_rope_apply<<<dim3((unsigned)((n + 255) / 256)), dim3(256), 0, stream>>>(
          XS, COS, SIN, QR, QRT);
    }
    k_gemm_scores<<<dim3((TPAD / 16) * (TPAD / 32) / 2, BATCH * NHEAD), dim3(64), 0, stream>>>(
        QR, QRT, SC);
    k_gemm_y<<<dim3((TPAD / 16) * (DMODEL / 64) / 4, BATCH * NHEAD), dim3(128), 0, stream>>>(
        SC, XN, Y);
    k_gemm_ysgate<<<dim3((TPAD / 16) * (NLAT / 64) / 4, BATCH * NHEAD), dim3(128), 0, stream>>>(
        Y, ENCV, XS, G);
    k_gemm_out<<<dim3((BT / 16) * (DMODEL / 64) / 4), dim3(128), 0, stream>>>(G, DEC, H);
  }

  // ---- final LN + mean-pool + head ----
  k_stats<<<dim3(BT), dim3(256), 0, stream>>>(H, ST);
  k_pool<<<dim3((BATCH * DMODEL + 255) / 256), dim3(256), 0, stream>>>(
      H, ST, lnf_w, lnf_b, POOL);
  k_head<<<dim3((BATCH * NCLS + 255) / 256), dim3(256), 0, stream>>>(
      POOL, head_w, head_b, out);
}
